// ConvolutionGGN_15908558865642
// MI455X (gfx1250) — matrix operand order repaired and checked by reading, not compiled, not run on hardware
//
#include <hip/hip_runtime.h>
#include <hip/hip_bf16.h>
#include <cstdint>
#include <cstddef>

typedef __attribute__((ext_vector_type(16))) _Float16 v16h;
typedef __attribute__((ext_vector_type(8)))  _Float16 v8h;
typedef __attribute__((ext_vector_type(4)))  _Float16 v4h;
typedef __attribute__((ext_vector_type(8)))  float    v8f;
typedef __attribute__((ext_vector_type(4)))  float    v4f;

namespace {
constexpr int   kN      = 1600;    // nodes
constexpr int   kE      = 38400;   // edges
constexpr int   kG      = 64;      // graphs
constexpr int   kNPG    = 25;      // nodes per graph
constexpr int   kFeat   = 2720;    // sum(H_DIMS[1:])
constexpr int   kMlpIn  = 5440;    // 2*feat
constexpr float kEps    = 1e-5f;
constexpr int   kMaxDeg = 48;
constexpr int   kChunks = 64;

__host__ __device__ inline int ceil_div(int a, int b) { return (a + b - 1) / b; }
} // namespace

// ---------------------------------------------------------------------------
// f32 -> f16 convert, 4 elements/thread (all our sizes are multiples of 4,
// but a scalar tail is kept for safety)
// ---------------------------------------------------------------------------
__global__ void k_cvt16(_Float16* __restrict__ dst, const float* __restrict__ src, int n) {
  int i4 = (blockIdx.x * 256 + threadIdx.x) * 4;
  if (i4 + 3 < n) {
    v4f v = *(const v4f*)(src + i4);
    v4h h;
#pragma unroll
    for (int t = 0; t < 4; ++t) h[t] = (_Float16)v[t];
    *(v4h*)(dst + i4) = h;
  } else {
    for (; i4 < n; ++i4) dst[i4] = (_Float16)src[i4];
  }
}

// ---------------------------------------------------------------------------
// Weight transpose + convert via LDS tile (coalesced read AND write):
//   wt[n*K + k] = (f16) w[k*N + n]
// ---------------------------------------------------------------------------
__global__ __launch_bounds__(256)
void k_wt16(_Float16* __restrict__ wt, const float* __restrict__ w, int K, int N) {
  __shared__ float tile[32][33];
  const int tx = threadIdx.x & 31;
  const int ty = threadIdx.x >> 5;     // 0..7
  const int kb = blockIdx.x << 5;
  const int nb = blockIdx.y << 5;
#pragma unroll
  for (int i = 0; i < 4; ++i) {
    int k = kb + ty + i * 8;
    int n = nb + tx;
    if (k < K && n < N) tile[ty + i * 8][tx] = w[(size_t)k * N + n];
  }
  __syncthreads();
#pragma unroll
  for (int i = 0; i < 4; ++i) {
    int n = nb + ty + i * 8;
    int k = kb + tx;
    if (n < N && k < K) wt[(size_t)n * K + k] = (_Float16)tile[tx][ty + i * 8];
  }
}

// ---------------------------------------------------------------------------
// WMMA GEMM: C[M x N] = A16[M x K] * Bt16[N x K]^T + bias (optional relu)
// One wave computes a 64 x (16*NT) tile. M % 64 == 0 for every call site
// (1600 / 38400 / 64); N edge handled by scalar NT dispatch so the K-loop is
// straight-line: 16 global_load_b128 -> 16 v_wmma, EXEC always all-ones.
// ---------------------------------------------------------------------------
template <int NT>
__device__ __forceinline__ void gemm_tile(
    const _Float16* __restrict__ A, const _Float16* __restrict__ Bt,
    const float* __restrict__ bias, float* __restrict__ C,
    int N, int K, int relu, int row0, int col0, int l16, int hf)
{
  v8f acc[4][NT];
#pragma unroll
  for (int mi = 0; mi < 4; ++mi)
#pragma unroll
    for (int ni = 0; ni < NT; ++ni)
#pragma unroll
      for (int e = 0; e < 8; ++e) acc[mi][ni][e] = 0.f;

  // A fragment (ISA 16-bit A 16x32): lane holds row l16; elems 0-15 at
  // K = k0 + hf*16 + s (contiguous, matching B).
  const _Float16* pa0 = A + (size_t)(row0 + l16) * K + hf * 16;
  // B fragment from pre-transposed weights: lane holds column l16,
  // 16 contiguous K values starting at k0 + hf*16.
  const _Float16* pb0 = Bt + (size_t)(col0 + l16) * K + hf * 16;

#pragma unroll 1
  for (int k0 = 0; k0 < K; k0 += 32) {
    v16h a[4], b[NT];
#pragma unroll
    for (int mi = 0; mi < 4; ++mi) {
      const _Float16* p = pa0 + (size_t)(mi * 16) * K + k0;
      v8h lo = *(const v8h*)p;
      v8h hi = *(const v8h*)(p + 8);
      a[mi] = __builtin_shufflevector(lo, hi, 0, 1, 2, 3, 4, 5, 6, 7,
                                      8, 9, 10, 11, 12, 13, 14, 15);
    }
#pragma unroll
    for (int ni = 0; ni < NT; ++ni)
      b[ni] = *(const v16h*)(pb0 + (size_t)(ni * 16) * K + k0);
#pragma unroll
    for (int mi = 0; mi < 4; ++mi)
#pragma unroll
      for (int ni = 0; ni < NT; ++ni)
        acc[mi][ni] = __builtin_amdgcn_wmma_f32_16x16x32_f16(
            false, a[mi], false, b[ni], (short)0, acc[mi][ni], false, false);
  }

  // C/D layout: lane = column (l16); VGPR r holds row hf*8 + r.
#pragma unroll
  for (int mi = 0; mi < 4; ++mi)
#pragma unroll
    for (int ni = 0; ni < NT; ++ni) {
      const int col = col0 + ni * 16 + l16;
      const float bv = bias ? bias[col] : 0.f;
#pragma unroll
      for (int r = 0; r < 8; ++r) {
        const int row = row0 + mi * 16 + hf * 8 + r;
        float v = acc[mi][ni][r] + bv;
        if (relu) v = fmaxf(v, 0.f);
        C[(size_t)row * N + col] = v;
      }
    }
}

__global__ __launch_bounds__(128)
void k_gemm_wmma(const _Float16* __restrict__ A, const _Float16* __restrict__ Bt,
                 const float* __restrict__ bias, float* __restrict__ C,
                 int M, int N, int K, int relu)
{
  const int lane = threadIdx.x & 31;
  const int tilesN = (N + 63) >> 6;
  const int tilesM = (M + 63) >> 6;
  // Scalarize the tile id (uniform per wave) so every guard below is an
  // SGPR compare + s_cbranch, never an exec mask (WMMA needs EXEC all-ones).
  const int id = __builtin_amdgcn_readfirstlane(blockIdx.x * 4 + (threadIdx.x >> 5));
  if (id >= tilesN * tilesM) return;
  const int tm = id / tilesN;
  const int tn = id - tm * tilesN;
  const int row0 = tm << 6;
  const int col0 = tn << 6;
  const int l16 = lane & 15;
  const int hf  = lane >> 4;
  const int rem = (N - col0) >> 4;   // valid 16-col subtiles at this col0 (scalar)
  if (rem >= 4)
    gemm_tile<4>(A, Bt, bias, C, N, K, relu, row0, col0, l16, hf);
  else if (rem == 3)
    gemm_tile<3>(A, Bt, bias, C, N, K, relu, row0, col0, l16, hf);
  else if (rem == 2)
    gemm_tile<2>(A, Bt, bias, C, N, K, relu, row0, col0, l16, hf);
  else
    gemm_tile<1>(A, Bt, bias, C, N, K, relu, row0, col0, l16, hf);
}

// ---------------------------------------------------------------------------
// Layer-0 edge embedding: E is (kE x 1)  ->  out[i,j] = E[i]*W[j] + b[j]
// ---------------------------------------------------------------------------
__global__ void k_outer(const float* __restrict__ E, const float* __restrict__ W,
                        const float* __restrict__ b, float* __restrict__ out, int h) {
  int idx = blockIdx.x * 256 + threadIdx.x;
  if (idx >= kE * h) return;
  int i = idx / h;
  int j = idx - i * h;
  out[idx] = E[i] * W[j] + b[j];
}

// e = CE + DX[src] + EX[dst]   (in place over CE)
__global__ void k_edge_gate(float* __restrict__ e, const float* __restrict__ DX,
                            const float* __restrict__ EX, const int* __restrict__ src,
                            const int* __restrict__ dst, int h) {
  int idx = blockIdx.x * 256 + threadIdx.x;
  if (idx >= kE * h) return;
  int k = idx / h;
  int j = idx - k * h;
  e[idx] += DX[(size_t)src[k] * h + j] + EX[(size_t)dst[k] * h + j];
}

// Build deterministic in-edge lists (fixed slot count) once per launch.
__global__ void k_build_in(const int* __restrict__ dst, int* __restrict__ inedge,
                           int* __restrict__ deg) {
  int n = blockIdx.x * 256 + threadIdx.x;
  if (n >= kN) return;
  int c = 0;
  for (int k = 0; k < kE; ++k)
    if (dst[k] == n && c < kMaxDeg) inedge[n * kMaxDeg + c++] = k;
  deg[n] = c;
}

// agg[n,j] = sum over in-edges k of n:  e[k,j] * BX[src[k],j]   (deterministic)
__global__ void k_agg(const float* __restrict__ e, const float* __restrict__ BX,
                      const int* __restrict__ src, const int* __restrict__ inedge,
                      const int* __restrict__ deg, float* __restrict__ agg, int h) {
  int idx = blockIdx.x * 256 + threadIdx.x;
  if (idx >= kN * h) return;
  int n = idx / h;
  int j = idx - n * h;
  float s = 0.f;
  const int d = deg[n];
  for (int r = 0; r < d; ++r) {
    int k = inedge[n * kMaxDeg + r];
    s += e[(size_t)k * h + j] * BX[(size_t)src[k] * h + j];
  }
  agg[idx] = s;
}

// out = relu(a + b)  (b may alias out)
__global__ void k_addrelu(float* __restrict__ out, const float* __restrict__ a,
                          const float* __restrict__ b, int n) {
  int i = blockIdx.x * 256 + threadIdx.x;
  if (i < n) out[i] = fmaxf(a[i] + b[i], 0.f);
}

// Column stats, pass 1: per-chunk partial sums (coalesced over columns).
__global__ void k_colsum_part(const float* __restrict__ X, int rows, int cols,
                              float* __restrict__ psum, float* __restrict__ psq) {
  int j = blockIdx.x * 256 + threadIdx.x;
  int c = blockIdx.y;
  if (j >= cols) return;
  int per = ceil_div(rows, kChunks);
  int r0 = c * per;
  int r1 = r0 + per; if (r1 > rows) r1 = rows;
  float s = 0.f, ss = 0.f;
  for (int r = r0; r < r1; ++r) {
    float v = X[(size_t)r * cols + j];
    s += v; ss += v * v;
  }
  psum[c * cols + j] = s;
  psq[c * cols + j]  = ss;
}

__global__ void k_colstat_fin(const float* __restrict__ psum, const float* __restrict__ psq,
                              int rows, int cols, float* __restrict__ mean,
                              float* __restrict__ var) {
  int j = blockIdx.x * 256 + threadIdx.x;
  if (j >= cols) return;
  float s = 0.f, ss = 0.f;
  for (int c = 0; c < kChunks; ++c) { s += psum[c * cols + j]; ss += psq[c * cols + j]; }
  float mu = s / (float)rows;
  mean[j] = mu;
  var[j]  = ss / (float)rows - mu * mu;   // biased variance (train-mode BN)
}

// out = relu(base + g*(x - mu)*rsqrt(var+eps) + b)
__global__ void k_bnres(float* __restrict__ out, const float* __restrict__ base,
                        const float* __restrict__ x, const float* __restrict__ mean,
                        const float* __restrict__ var, const float* __restrict__ g,
                        const float* __restrict__ b, int total, int cols) {
  int idx = blockIdx.x * 256 + threadIdx.x;
  if (idx >= total) return;
  int j = idx % cols;
  float xn = g[j] * (x[idx] - mean[j]) * rsqrtf(var[j] + kEps) + b[j];
  out[idx] = fmaxf(base[idx] + xn, 0.f);
}

// parents = mean over rows 0,2,...,22; children = mean over rows 1,3,...,23
__global__ void k_pool(const float* __restrict__ H, int d, float* __restrict__ pm,
                       float* __restrict__ cm, int foff) {
  int idx = blockIdx.x * 256 + threadIdx.x;
  if (idx >= kG * d) return;
  int g = idx / d;
  int j = idx - g * d;
  const float* Hg = H + (size_t)g * kNPG * d + j;
  float sp = 0.f, sc = 0.f;
#pragma unroll
  for (int r = 0; r < 12; ++r) {
    sp += Hg[(size_t)(2 * r) * d];
    sc += Hg[(size_t)(2 * r + 1) * d];
  }
  pm[(size_t)g * kFeat + foff + j] = sp * (1.f / 12.f);
  cm[(size_t)g * kFeat + foff + j] = sc * (1.f / 12.f);
}

__global__ void k_concat(float* __restrict__ mlp_in, const float* __restrict__ pm,
                         const float* __restrict__ cm) {
  int idx = blockIdx.x * 256 + threadIdx.x;
  if (idx >= kG * kMlpIn) return;
  int g = idx / kMlpIn;
  int j = idx - g * kMlpIn;
  mlp_in[idx] = (j < kFeat) ? pm[(size_t)g * kFeat + j]
                            : cm[(size_t)g * kFeat + (j - kFeat)];
}

// y[g,o] = center[g,:] . w2[:,o] + b2[o]   (N=2 output head, fp32)
__global__ void k_final(float* __restrict__ y, const float* __restrict__ center,
                        const float* __restrict__ w2, const float* __restrict__ b2) {
  int idx = blockIdx.x * 128 + threadIdx.x;
  if (idx >= kG * 2) return;
  int g = idx >> 1;
  int o = idx & 1;
  float s = b2[o];
  for (int k = 0; k < kFeat; ++k) s += center[(size_t)g * kFeat + k] * w2[k * 2 + o];
  y[idx] = s;
}

// ---------------------------------------------------------------------------
// Host orchestration
// ---------------------------------------------------------------------------
static void cvt16(hipStream_t s, _Float16* dst, const float* src, int n) {
  k_cvt16<<<ceil_div(ceil_div(n, 4), 256), 256, 0, s>>>(dst, src, n);
}
static void wt16(hipStream_t s, _Float16* wt, const float* w, int K, int N) {
  dim3 grid(ceil_div(K, 32), ceil_div(N, 32));
  k_wt16<<<grid, 256, 0, s>>>(wt, w, K, N);
}
static void gemm(hipStream_t s, const _Float16* A, const _Float16* Bt, const float* bias,
                 float* C, int M, int N, int K, int relu) {
  int tiles = ceil_div(M, 64) * ceil_div(N, 64);
  k_gemm_wmma<<<ceil_div(tiles, 4), 128, 0, s>>>(A, Bt, bias, C, M, N, K, relu);
}
static void colstats(hipStream_t s, const float* X, int rows, int cols,
                     float* psum, float* psq, float* mean, float* var) {
  dim3 grid(ceil_div(cols, 256), kChunks);
  k_colsum_part<<<grid, 256, 0, s>>>(X, rows, cols, psum, psq);
  k_colstat_fin<<<ceil_div(cols, 256), 256, 0, s>>>(psum, psq, rows, cols, mean, var);
}

extern "C" void kernel_launch(void* const* d_in, const int* in_sizes, int n_in,
                              void* d_out, int out_size, void* d_ws, size_t ws_size,
                              hipStream_t stream) {
  (void)in_sizes; (void)n_in; (void)out_size; (void)ws_size;

  // Input layout: dict insertion order of setup_inputs():
  //   0:X  1:E  2..73: params[0..3] x 18 each  74..77: mlp  78:src 79:dst 80:batch
  const float* X    = (const float*)d_in[0];
  const float* Ein0 = (const float*)d_in[1];
  const float* w1   = (const float*)d_in[74];
  const float* b1   = (const float*)d_in[75];
  const float* w2   = (const float*)d_in[76];
  const float* b2   = (const float*)d_in[77];
  const int*   src  = (const int*)d_in[78];
  const int*   dst  = (const int*)d_in[79];
  auto P = [&](int L, int off) { return (const float*)d_in[2 + L * 18 + off]; };
  // per-layer param offsets: 0:emb_h_w 1:emb_h_b 2:emb_e_w 3:emb_e_b 4:A_w 5:A_b
  // 6:B_w 7:B_b 8:C_w 9:C_b 10:D_w 11:D_b 12:Ee_w 13:Ee_b 14:bn_h_g 15:bn_h_b
  // 16:bn_e_g 17:bn_e_b

  // Workspace carve (~1.25 GB total)
  char* wsp = (char*)d_ws;
  auto carve = [&](size_t bytes) -> void* {
    void* p = (void*)wsp;
    wsp += (bytes + 255) & ~(size_t)255;
    return p;
  };
  float* pEa   = (float*)carve((size_t)kE * 2048 * 4);  // Eemb
  float* pEb   = (float*)carve((size_t)kE * 2048 * 4);  // CE -> e
  float* pEc   = (float*)carve((size_t)kE * 2048 * 4);  // Eout / next Ein
  float* pHemb = (float*)carve((size_t)kN * 2048 * 4);
  float* pHin  = (float*)carve((size_t)kN * 2048 * 4);  // layer output H
  float* pAX   = (float*)carve((size_t)kN * 2048 * 4);
  float* pBX   = (float*)carve((size_t)kN * 2048 * 4);
  float* pDX   = (float*)carve((size_t)kN * 2048 * 4);
  float* pEX   = (float*)carve((size_t)kN * 2048 * 4);
  float* pAgg  = (float*)carve((size_t)kN * 2048 * 4);
  _Float16* act16 = (_Float16*)carve((size_t)kE * 2048 * 2);
  _Float16* w16   = (_Float16*)carve((size_t)kFeat * kMlpIn * 2);
  float* mlp_in = (float*)carve((size_t)kG * kMlpIn * 4);
  float* psum   = (float*)carve((size_t)kChunks * 2048 * 4);
  float* psq    = (float*)carve((size_t)kChunks * 2048 * 4);
  float* mean   = (float*)carve(2048 * 4);
  float* var    = (float*)carve(2048 * 4);
  int*   inedge = (int*)carve((size_t)kN * kMaxDeg * 4);
  int*   deg    = (int*)carve((size_t)kN * 4);

  // Output layout: y (64x2) | pm (64x2720) | cm (64x2720) | center (64x2720)
  float* y      = (float*)d_out;
  float* pm     = y + kG * 2;
  float* cm     = pm + (size_t)kG * kFeat;
  float* center = cm + (size_t)kG * kFeat;

  // Deterministic in-edge lists (shared by all 4 layers)
  k_build_in<<<ceil_div(kN, 256), 256, 0, stream>>>(dst, inedge, deg);

  const int dins[4]  = {2048, 2048, 512, 128};
  const int hs[4]    = {2048, 512, 128, 32};
  const int edins[4] = {1, 2048, 512, 128};
  const int foff[4]  = {0, 2048, 2560, 2688};

  const float* Hsrc = X;
  for (int L = 0; L < 4; ++L) {
    const int din = dins[L], h = hs[L], edin = edins[L];

    // H = Hin @ emb_h_w + b
    cvt16(stream, act16, Hsrc, kN * din);
    wt16(stream, w16, P(L, 0), din, h);
    gemm(stream, act16, w16, P(L, 1), pHemb, kN, h, din, 0);

    // E = Ein @ emb_e_w + b  (layer0: K=1 outer product)
    if (L == 0) {
      k_outer<<<ceil_div(kE * h, 256), 256, 0, stream>>>(Ein0, P(0, 2), P(0, 3), pEa, h);
    } else {
      cvt16(stream, act16, pEc, kE * edin);
      wt16(stream, w16, P(L, 2), edin, h);
      gemm(stream, act16, w16, P(L, 3), pEa, kE, h, edin, 0);
    }

    // AX, BX, DX, EX from Hemb
    cvt16(stream, act16, pHemb, kN * h);
    wt16(stream, w16, P(L, 4), h, h);
    gemm(stream, act16, w16, P(L, 5), pAX, kN, h, h, 0);
    wt16(stream, w16, P(L, 6), h, h);
    gemm(stream, act16, w16, P(L, 7), pBX, kN, h, h, 0);
    wt16(stream, w16, P(L, 10), h, h);
    gemm(stream, act16, w16, P(L, 11), pDX, kN, h, h, 0);
    wt16(stream, w16, P(L, 12), h, h);
    gemm(stream, act16, w16, P(L, 13), pEX, kN, h, h, 0);

    // CE = Eemb @ C_w + b   (the big one)
    cvt16(stream, act16, pEa, kE * h);
    wt16(stream, w16, P(L, 8), h, h);
    gemm(stream, act16, w16, P(L, 9), pEb, kE, h, h, 0);

    // e = CE + DX[src] + EX[dst]     (in place over CE)
    k_edge_gate<<<ceil_div(kE * h, 256), 256, 0, stream>>>(pEb, pDX, pEX, src, dst, h);

    // agg = segment_sum(e * BX[src], dst); hpre = relu(AX + agg)
    k_agg<<<ceil_div(kN * h, 256), 256, 0, stream>>>(pEb, pBX, src, inedge, deg, pAgg, h);
    k_addrelu<<<ceil_div(kN * h, 256), 256, 0, stream>>>(pAgg, pAX, pAgg, kN * h);

    // Hout = relu(Hemb + BN(hpre))
    colstats(stream, pAgg, kN, h, psum, psq, mean, var);
    k_bnres<<<ceil_div(kN * h, 256), 256, 0, stream>>>(pHin, pHemb, pAgg, mean, var,
                                                       P(L, 14), P(L, 15), kN * h, h);

    // Eout = relu(Eemb + BN(e))   -> pEc (next layer's Ein)
    colstats(stream, pEb, kE, h, psum, psq, mean, var);
    k_bnres<<<ceil_div(kE * h, 256), 256, 0, stream>>>(pEc, pEa, pEb, mean, var,
                                                       P(L, 16), P(L, 17), kE * h, h);

    // pooling into pm/cm feature slices
    k_pool<<<ceil_div(kG * h, 256), 256, 0, stream>>>(pHin, h, pm, cm, foff[L]);

    Hsrc = pHin;
  }

  // MLP head: center = relu([pm|cm] @ w1 + b1); y = center @ w2 + b2
  k_concat<<<ceil_div(kG * kMlpIn, 256), 256, 0, stream>>>(mlp_in, pm, cm);
  cvt16(stream, act16, mlp_in, kG * kMlpIn);
  wt16(stream, w16, w1, kMlpIn, kFeat);
  gemm(stream, act16, w16, b1, center, kG, kFeat, kMlpIn, 1);
  k_final<<<1, 128, 0, stream>>>(y, center, w2, b2);
}
